// BasicBlock_3865470566930
// MI455X (gfx1250) — matrix operand order repaired and checked by reading, not compiled, not run on hardware
//
#include <hip/hip_runtime.h>
#include <hip/hip_bf16.h>
#include <math.h>

typedef __attribute__((ext_vector_type(16))) _Float16 v16h;
typedef __attribute__((ext_vector_type(8)))  _Float16 v8h;
typedef __attribute__((ext_vector_type(8)))  float    v8f;

#define CDIM 256
#define KNBR 9
#define BM   64
#define LDA  264   // halves per staged row (256 + 8 pad, keeps 16B align, kills bank conflicts)
#define LDF  260   // floats per epilogue row (256 + 4 pad)

__device__ __forceinline__ float gelu_exact(float x) {
    return 0.5f * x * (1.0f + erff(x * 0.70710678118654752f));
}

// ---- f32 -> f16 conversion of the feature table (done once; 9 gathers reuse it) ----
__global__ void cvt_f32_to_f16(const float* __restrict__ src,
                               _Float16* __restrict__ dst, int n) {
    int i = blockIdx.x * blockDim.x + threadIdx.x;
    if (i < n) dst[i] = (_Float16)src[i];
}

// ---- Pack W [9,256,256] f32 into WMMA-B fragment order (f16) ----
// Linear dest index d = (((k*8 + cc)*16 + nt)*32 + lane)*16 + i
// Fragment semantics (B is 32x16, KxN):
//   lanes 0-15 hold K = 0..15 (N = lane), lanes 16-31 hold K = 16..31 (N = lane-16)
//   element i of the per-lane v16h is K = (lane>>4)*16 + i within the cc-th 32-chunk.
__global__ void pack_w(const float* __restrict__ W, _Float16* __restrict__ Wp, int total) {
    int d = blockIdx.x * blockDim.x + threadIdx.x;
    if (d >= total) return;
    int i  = d & 15;
    int L  = (d >> 4) & 31;
    int nt = (d >> 9) & 15;
    int cc = (d >> 13) & 7;
    int k  = d >> 16;
    int kk = cc * 32 + (L >> 4) * 16 + i;    // K index within [0,256)
    int n  = nt * 16 + (L & 15);             // output channel
    Wp[d] = (_Float16)W[((size_t)k * CDIM + kk) * CDIM + n];
}

// ---- fused gather-GEMM (WMMA) + LayerNorm (+residual) + GELU ----
// PASS 1: out = gelu(LN1(conv1(feats)))  -> f16 workspace
// PASS 2: out = gelu(LN2(conv2(h)) + feats) -> f32 d_out
template<int PASS>
__global__ void __launch_bounds__(256)
spconv_block(const _Float16* __restrict__ X,     // gather source, f16 [N,256]
             const _Float16* __restrict__ Wp,    // packed weights, f16
             const float* __restrict__ gamma,
             const float* __restrict__ beta,
             const int* __restrict__ nbr,
             const unsigned char* __restrict__ mask,
             const float* __restrict__ resid,    // PASS 2 only
             _Float16* __restrict__ outH,        // PASS 1 output
             float* __restrict__ outF,           // PASS 2 output
             int N)
{
    __shared__ __align__(16) unsigned char smem_raw[BM * LDF * 4];
    _Float16* smA = (_Float16*)smem_raw;   // staged gathered tile  [BM][LDA] f16
    float*    smF = (float*)smem_raw;      // epilogue result tile  [BM][LDF] f32

    const int tid  = threadIdx.x;
    const int lane = tid & 31;
    const int wave = tid >> 5;
    const int mla  = lane & 15;            // M within 16 (A/C layout)
    const int hse  = lane >> 4;            // half-select (K/M-half)
    const int mt   = wave >> 1;            // wave's M-tile (0..3)
    const int nt0  = (wave & 1) * 8;       // wave's first N-tile (0 or 8)
    const int n0   = blockIdx.x * BM;

    v8f acc[8] = {};                       // 8 16x16 f32 accumulators per wave

    for (int k = 0; k < KNBR; ++k) {
        // -- stage 64 gathered (masked) f16 rows into LDS, 16B segments, coalesced --
        #pragma unroll 1
        for (int s = tid; s < BM * 32; s += 256) {
            int row = s >> 5, seg = s & 31;
            int node = n0 + row;
            v8h val = {};
            if (node < N) {
                int base = node * KNBR + k;
                if (mask[base]) {
                    int nb = nbr[base];
                    val = *(const v8h*)(X + (size_t)nb * CDIM + seg * 8);
                }
            }
            *(v8h*)(smA + row * LDA + seg * 8) = val;
        }
        __syncthreads();

        const _Float16* WpK = Wp + (size_t)k * (8 * 16 * 512);
        #pragma unroll
        for (int cc = 0; cc < 8; ++cc) {
            // A fragment (contiguous K order, matching packed B):
            // lane holds row M=mla; elems 0..15 -> K = cc*32 + hse*16 + 0..15
            const _Float16* ap = smA + (mt * 16 + mla) * LDA + cc * 32 + hse * 16;
            v8h lo = *(const v8h*)(ap);
            v8h hi = *(const v8h*)(ap + 8);
            v16h a = __builtin_shufflevector(lo, hi,
                        0,1,2,3,4,5,6,7,8,9,10,11,12,13,14,15);
            const _Float16* bp = WpK + ((size_t)cc * 16 + nt0) * 512 + lane * 16;
            #pragma unroll
            for (int j = 0; j < 8; ++j) {
                v16h b = *(const v16h*)(bp + j * 512);   // 32B/lane, 1KB/wave, contiguous
                acc[j] = __builtin_amdgcn_wmma_f32_16x16x32_f16(
                            false, a, false, b, (short)0, acc[j], false, false);
            }
        }
        __syncthreads();   // protect smA before next k overwrites it
    }

    // -- spill accumulators to LDS (C/D layout: VGPR v -> M = v + 8*hse) --
    #pragma unroll
    for (int j = 0; j < 8; ++j) {
        int col = (nt0 + j) * 16 + mla;
        #pragma unroll
        for (int v = 0; v < 8; ++v) {
            int row = mt * 16 + hse * 8 + v;
            smF[row * LDF + col] = acc[j][v];
        }
    }
    __syncthreads();

    // -- LayerNorm: 4 lanes (a quad) per node, quad shuffle reduction --
    int nodeL = tid >> 2;
    int q     = tid & 3;
    const float* rowp = smF + nodeL * LDF;
    float sum = 0.f, sq = 0.f;
    #pragma unroll
    for (int i = 0; i < 64; ++i) {
        float x = rowp[q + 4 * i];     // interleaved -> conflict-free banks
        sum += x; sq += x * x;
    }
    sum += __shfl_xor(sum, 1, 32); sum += __shfl_xor(sum, 2, 32);
    sq  += __shfl_xor(sq,  1, 32); sq  += __shfl_xor(sq,  2, 32);
    float mu  = sum * (1.0f / 256.0f);
    float var = sq  * (1.0f / 256.0f) - mu * mu;
    float inv = rsqrtf(var + 1e-6f);

    int node = n0 + nodeL;
    if (node < N) {
        #pragma unroll 1
        for (int i = 0; i < 64; ++i) {
            int c = q + 4 * i;
            float x = (rowp[c] - mu) * inv * gamma[c] + beta[c];
            if (PASS == 1) {
                outH[(size_t)node * CDIM + c] = (_Float16)gelu_exact(x);
            } else {
                float y = x + resid[(size_t)node * CDIM + c];
                outF[(size_t)node * CDIM + c] = gelu_exact(y);
            }
        }
    }
}

extern "C" void kernel_launch(void* const* d_in, const int* in_sizes, int n_in,
                              void* d_out, int out_size, void* d_ws, size_t ws_size,
                              hipStream_t stream) {
    const float* feats = (const float*)d_in[0];
    const float* W1    = (const float*)d_in[1];
    const float* g1    = (const float*)d_in[2];
    const float* b1    = (const float*)d_in[3];
    const float* W2    = (const float*)d_in[4];
    const float* g2    = (const float*)d_in[5];
    const float* b2    = (const float*)d_in[6];
    const int*   nbr   = (const int*)d_in[7];
    const unsigned char* mask = (const unsigned char*)d_in[8];
    float* out = (float*)d_out;

    int N = in_sizes[0] / CDIM;
    size_t feat_elems = (size_t)N * CDIM;

    // workspace layout (f16): feats_h | h_h | Wp1 | Wp2   (~105 MB total)
    _Float16* feats_h = (_Float16*)d_ws;
    _Float16* h_h     = feats_h + feat_elems;
    _Float16* Wp1     = h_h + feat_elems;
    _Float16* Wp2     = Wp1 + (size_t)KNBR * 65536;

    const int threads = 256;
    cvt_f32_to_f16<<<(int)((feat_elems + 255) / 256), threads, 0, stream>>>(
        feats, feats_h, (int)feat_elems);

    int wtotal = KNBR * 65536;   // 589,824 f16 elems per weight tensor
    pack_w<<<(wtotal + 255) / 256, threads, 0, stream>>>(W1, Wp1, wtotal);
    pack_w<<<(wtotal + 255) / 256, threads, 0, stream>>>(W2, Wp2, wtotal);

    int nblk = (N + BM - 1) / BM;
    spconv_block<1><<<nblk, threads, 0, stream>>>(feats_h, Wp1, g1, b1, nbr, mask,
                                                  nullptr, h_h, nullptr, N);
    spconv_block<2><<<nblk, threads, 0, stream>>>(h_h, Wp2, g2, b2, nbr, mask,
                                                  feats, nullptr, out, N);
}
